// PPCA_41231686041776
// MI455X (gfx1250) — matrix operand order repaired and checked by reading, not compiled, not run on hardware
//
#include <hip/hip_runtime.h>

// ---------------------------------------------------------------------------
// PPCA E-step on gfx1250 (MI455X), wave32.
//
// Per 16-sample group (one wave):
//   Oww[16 x 256]  = O[16 x 128](bf16 exact) x G[128 x 256](bf16 hi+lo)  [WMMA]
//   proj[16 x 16]  = t[16 x 128](bf16 hi+lo) x W[128 x 16](bf16 hi+lo)  [WMMA]
//   Sigma          = vy * inv(vy I + Oww)   (in-place Gauss-Jordan, 2 samples
//                     per wave, ds_bpermute broadcasts; SPD -> no pivoting)
//   xbar           = inv @ proj   (transposed stride-17 LDS scratch, no
//                     butterfly reductions)
// G = { w_i w_i^T } is precomputed per block into LDS, already permuted into
// a 16-bit B-operand lane layout with contiguous k order (k = 16*(lane/16) + e) so each lane loads 32B contiguous.
// WMMA uses 4 independent accumulator chains (2 tiles x hi/lo) to avoid the
// bf16 WMMA->WMMA RAW hazard bubbles.
// ---------------------------------------------------------------------------

typedef __attribute__((ext_vector_type(16))) __bf16       v16bf;
typedef __attribute__((ext_vector_type(8)))  float        v8f;
typedef __attribute__((ext_vector_type(8)))  unsigned int v8u;

#define NW   6          // waves per block (one block per WGP; LDS-limited)
#define WSZ  32

// ---- LDS carve (bytes), all 32B aligned ----
#define OFF_M    0
#define SZ_M     512                      // m: 128 f32
#define OFF_WH   (OFF_M + SZ_M)           // W hi (staged B layout): 4*512 u16
#define SZ_W     4096
#define OFF_WL   (OFF_WH + SZ_W)
#define OFF_GH   (OFF_WL + SZ_W)          // G hi (staged B layout): 4*16*512 u16
#define SZ_G     65536
#define OFF_GL   (OFF_GH + SZ_G)
#define OFF_PW   (OFF_GL + SZ_G)          // per-wave region below
#define OM_STR   144                      // mask row stride (bytes): 16B & bank friendly
#define YS_STR   132                      // Y row stride (floats): 16B & bank friendly
#define SIG_STR  276                      // Sigma row stride (floats): bank friendly
#define PW_OM    (16*OM_STR)              // 2304
#define PW_YS    (16*YS_STR*4)            // 8448 (reused as xbar scratch in GJ phase)
#define PW_SIG   (16*SIG_STR*4)           // 17664
#define PW_PJ    1024                     // proj: 16x16 f32
#define PW_BYTES (PW_OM+PW_YS+PW_SIG+PW_PJ)      // 29440
#define SMEM_BYTES (OFF_PW + NW*PW_BYTES)        // 316416 (< 320KB/WGP)

static __device__ __forceinline__ unsigned short bf16_rne(float f) {
  unsigned u = __builtin_bit_cast(unsigned, f);
  unsigned r = u + 0x7FFFu + ((u >> 16) & 1u);
  return (unsigned short)(r >> 16);
}
static __device__ __forceinline__ float bf16_f32(unsigned short h) {
  unsigned u = ((unsigned)h) << 16;
  return __builtin_bit_cast(float, u);
}
static __device__ __forceinline__ v16bf lds_b16x16(const unsigned short* p) {
  return __builtin_bit_cast(v16bf, *(const v8u*)p);   // 2x ds_load_b128
}
static __device__ __forceinline__ float bperm(int src_lane_x4, float v) {
  return __builtin_bit_cast(float,
      __builtin_amdgcn_ds_bpermute(src_lane_x4, __builtin_bit_cast(int, v)));
}
// 16-bit A-operand element e -> K index (contiguous order, matching staged B),
// g = lane>>4:  k = 16*g + e
static __device__ __forceinline__ int a_e2k(int e, int g) {
  return (g << 4) + e;
}
#define WMMA_BF16(A, B, C) \
  __builtin_amdgcn_wmma_f32_16x16x32_bf16(false, (A), false, (B), (short)0, (C), false, false)

__global__ void __launch_bounds__(NW * WSZ, 1)
ppca_estep(const float* __restrict__ Y, const unsigned char* __restrict__ O,
           const float* __restrict__ mvec, const float* __restrict__ Wmat,
           const float* __restrict__ vyp, float* __restrict__ outX,
           float* __restrict__ outS, int Ntot, int ngroups) {
  extern __shared__ unsigned char smem[];
  float*          mS    = (float*)(smem + OFF_M);
  unsigned short* WstHi = (unsigned short*)(smem + OFF_WH);
  unsigned short* WstLo = (unsigned short*)(smem + OFF_WL);
  unsigned short* GstHi = (unsigned short*)(smem + OFF_GH);
  unsigned short* GstLo = (unsigned short*)(smem + OFF_GL);

  const int tid  = threadIdx.x;
  const int wave = tid >> 5;
  const int lane = tid & 31;
  const int g16  = lane >> 4;       // which 16-lane half
  const int col  = lane & 15;

  unsigned char* pw   = smem + OFF_PW + wave * PW_BYTES;
  unsigned char* OmW  = pw;
  float*         YsW  = (float*)(pw + PW_OM);
  float*         SigW = (float*)(pw + PW_OM + PW_YS);
  float*         PjW  = (float*)(pw + PW_OM + PW_YS + PW_SIG);
  float*         XTW  = YsW;        // xbar scratch aliases Ys (Ys dead by then)

  // -------- one-time per-block setup: m, W(hi/lo), G(hi/lo) staged in LDS ----
  for (int i = tid; i < 128; i += NW * WSZ) mS[i] = mvec[i];

  // W staged into 16-bit B-operand layout: value W[kc*32+k][n] at
  //   lane = n + 16*(k>>4), e = k&15
  for (int idx = tid; idx < 128 * 16; idx += NW * WSZ) {
    int i = idx >> 4, n = idx & 15;
    float v = Wmat[i * 16 + n];
    unsigned short hi = bf16_rne(v);
    unsigned short lo = bf16_rne(v - bf16_f32(hi));
    int kc = i >> 5, kk = i & 31;
    int pos = (kc << 9) + ((n + ((kk >> 4) << 4)) << 4) + (kk & 15);
    WstHi[pos] = hi; WstLo[pos] = lo;
  }
  // G[i][col=(k,l)] = w[i][k]*w[i][l], staged per (kc,tcol) tile in B layout.
  for (int idx = tid; idx < 128 * 256; idx += NW * WSZ) {
    int i = idx >> 8, c = idx & 255;
    float gv = Wmat[i * 16 + (c >> 4)] * Wmat[i * 16 + (c & 15)];
    unsigned short hi = bf16_rne(gv);
    unsigned short lo = bf16_rne(gv - bf16_f32(hi));
    int kc = i >> 5, kk = i & 31;
    int pos = ((kc * 16 + (c >> 4)) << 9) +
              (((c & 15) + ((kk >> 4) << 4)) << 4) + (kk & 15);
    GstHi[pos] = hi; GstLo[pos] = lo;
  }
  __syncthreads();

  const float vy = vyp[0];
  const int g0 = blockIdx.x * NW + wave;
  const int gstride = gridDim.x * NW;

  for (int grp = g0; grp < ngroups; grp += gstride) {
    const int n0 = grp * 16;
    // prefetch next group's inputs (global_prefetch_b8)
    int gn = grp + gstride;
    if (gn < ngroups) {
      __builtin_prefetch(Y + (size_t)gn * 16 * 128, 0, 0);
      __builtin_prefetch(O + (size_t)gn * 16 * 128, 0, 0);
    }

    // ---- cooperative, coalesced global->LDS copies (wave-private, padded) ---
#pragma unroll
    for (int t = 0; t < 16; ++t) {              // Y: 16 rows x 32 float4
      int f = lane + (t << 5);
      int row = f >> 5, c4 = f & 31;
      int nr = n0 + row; if (nr >= Ntot) nr = Ntot - 1;
      const float4* src = (const float4*)(Y + (size_t)nr * 128);
      ((float4*)(YsW + row * YS_STR))[c4] = src[c4];
    }
#pragma unroll
    for (int t = 0; t < 4; ++t) {               // O: 16 rows x 8 uint4
      int f = lane + (t << 5);
      int row = f >> 3, c4 = f & 7;
      int nr = n0 + row; if (nr >= Ntot) nr = Ntot - 1;
      const uint4* src = (const uint4*)(O + (size_t)nr * 128);
      ((uint4*)(OmW + row * OM_STR))[c4] = src[c4];
    }
    asm volatile("s_wait_loadcnt 0\n\ts_wait_dscnt 0" ::: "memory");

    // ---- build mask A-operands (bf16 is exact for 0/1): AO[kc] -------------
    v16bf AO[4];
#pragma unroll
    for (int kc = 0; kc < 4; ++kc) {
      v16bf a = __builtin_bit_cast(v16bf, (v8u)0);
#pragma unroll
      for (int e = 0; e < 16; ++e) {
        int i = (kc << 5) + a_e2k(e, g16);
        unsigned short h = OmW[col * OM_STR + i] ? (unsigned short)0x3F80
                                                 : (unsigned short)0;
        a[e] = __builtin_bit_cast(__bf16, h);
      }
      AO[kc] = a;
    }

    // ---- Oww: 16 output tiles, K=128, hi+lo split B.  Two tiles per pass
    //      with separate hi/lo accumulators -> 4 independent WMMA chains
    //      (no WMMA->WMMA RAW bubbles). -----------------------------------
#pragma unroll 1
    for (int tp = 0; tp < 8; ++tp) {
      const int t0 = 2 * tp, t1 = 2 * tp + 1;
      v8f aH0 = {0.f,0.f,0.f,0.f,0.f,0.f,0.f,0.f};
      v8f aL0 = {0.f,0.f,0.f,0.f,0.f,0.f,0.f,0.f};
      v8f aH1 = {0.f,0.f,0.f,0.f,0.f,0.f,0.f,0.f};
      v8f aL1 = {0.f,0.f,0.f,0.f,0.f,0.f,0.f,0.f};
#pragma unroll
      for (int kc = 0; kc < 4; ++kc) {
        const int b0 = ((kc * 16 + t0) << 9) + (lane << 4);
        const int b1 = ((kc * 16 + t1) << 9) + (lane << 4);
        v16bf bh0 = lds_b16x16(GstHi + b0);
        v16bf bh1 = lds_b16x16(GstHi + b1);
        v16bf bl0 = lds_b16x16(GstLo + b0);
        v16bf bl1 = lds_b16x16(GstLo + b1);
        aH0 = WMMA_BF16(AO[kc], bh0, aH0);
        aH1 = WMMA_BF16(AO[kc], bh1, aH1);
        aL0 = WMMA_BF16(AO[kc], bl0, aL0);
        aL1 = WMMA_BF16(AO[kc], bl1, aL1);
      }
      // dump tiles to SigW with +vy on diagonal (entry (k=tcol,l=n), diag n==tcol)
      float d0 = (col == t0) ? vy : 0.f;
      float d1 = (col == t1) ? vy : 0.f;
#pragma unroll
      for (int r = 0; r < 8; ++r) {
        int s = r + (g16 << 3);
        SigW[s * SIG_STR + t0 * 16 + col] = (aH0[r] + aL0[r]) + d0;
        SigW[s * SIG_STR + t1 * 16 + col] = (aH1[r] + aL1[r]) + d1;
      }
    }

    // ---- proj = t x W with hi/lo split on both sides: 3 independent chains --
    {
      v8f pa = {0.f,0.f,0.f,0.f,0.f,0.f,0.f,0.f};
      v8f pb = {0.f,0.f,0.f,0.f,0.f,0.f,0.f,0.f};
      v8f pc = {0.f,0.f,0.f,0.f,0.f,0.f,0.f,0.f};
#pragma unroll
      for (int kc = 0; kc < 4; ++kc) {
        v16bf ah = __builtin_bit_cast(v16bf, (v8u)0);
        v16bf al = __builtin_bit_cast(v16bf, (v8u)0);
#pragma unroll
        for (int e = 0; e < 16; ++e) {
          int i = (kc << 5) + a_e2k(e, g16);
          float y = YsW[col * YS_STR + i];
          y = (y == y) ? y : 0.f;                       // nan_to_num
          float tv = OmW[col * OM_STR + i] ? (y - mS[i]) : 0.f;
          unsigned short hh = bf16_rne(tv);
          unsigned short hl = bf16_rne(tv - bf16_f32(hh));
          ah[e] = __builtin_bit_cast(__bf16, hh);
          al[e] = __builtin_bit_cast(__bf16, hl);
        }
        v16bf bwh = lds_b16x16(WstHi + (kc << 9) + (lane << 4));
        v16bf bwl = lds_b16x16(WstLo + (kc << 9) + (lane << 4));
        pa = WMMA_BF16(ah, bwh, pa);
        pb = WMMA_BF16(ah, bwl, pb);
        pc = WMMA_BF16(al, bwh, pc);
      }
#pragma unroll
      for (int r = 0; r < 8; ++r)
        PjW[(r + (g16 << 3)) * 16 + col] = (pa[r] + pb[r]) + pc[r];
    }
    asm volatile("s_wait_dscnt 0" ::: "memory");

    // ---- per-sample 16x16 SPD inverse + xbar: 2 samples per wave -----------
#pragma unroll 1
    for (int q = 0; q < 8; ++q) {
      const int s = 2 * q + g16;                 // this half's sample
      float inv[16];
#pragma unroll
      for (int r = 0; r < 16; ++r)
        inv[r] = SigW[s * SIG_STR + r * 16 + col];

      const int hbase = (lane & 16) << 2;        // byte lane base of this half
      // in-place Gauss-Jordan (no pivoting: matrix is vy*I + PSD).
      // Column-k lanes: rowk==ip, C-operand forced to 0 -> fma yields -f*ip.
#pragma unroll
      for (int k = 0; k < 16; ++k) {
        const int bidx = hbase + (k << 2);
        float piv = bperm(bidx, inv[k]);
        float ip = 1.0f / piv;
        const bool ck = (col == k);
        float rowk = ck ? ip : inv[k] * ip;
        inv[k] = rowk;
#pragma unroll
        for (int j = 0; j < 16; ++j) {
          if (j == k) continue;
          float f = bperm(bidx, inv[j]);
          inv[j] = fmaf(-f, rowk, ck ? 0.f : inv[j]);
        }
      }

      // stash inv transposed (stride 17: conflict-free) for the row-dot xbar
#pragma unroll
      for (int r = 0; r < 16; ++r)
        XTW[(g16 << 8) + (g16 << 4) + col * 17 + r] = inv[r];  // g16*272
      asm volatile("s_wait_dscnt 0" ::: "memory");

      // xbar[s][col] = sum_c inv[col][c] * proj[s][c]
      float xb = 0.f;
#pragma unroll
      for (int c = 0; c < 16; ++c)
        xb = fmaf(XTW[(g16 << 8) + (g16 << 4) + c * 17 + col],
                  PjW[s * 16 + c], xb);

      if (n0 + s < Ntot) {
        outX[(size_t)(n0 + s) * 16 + col] = xb;
        float* so = outS + (size_t)(n0 + s) * 256 + col;
#pragma unroll
        for (int r = 0; r < 16; ++r) so[r * 16] = vy * inv[r];  // Sigma = vy*inv
      }
    }
  }
}

extern "C" void kernel_launch(void* const* d_in, const int* in_sizes, int n_in,
                              void* d_out, int out_size, void* d_ws, size_t ws_size,
                              hipStream_t stream) {
  (void)n_in; (void)out_size; (void)d_ws; (void)ws_size;
  const float*         Y  = (const float*)d_in[0];
  const unsigned char* O  = (const unsigned char*)d_in[1];  // jnp bool_ = 1 byte
  const float*         m  = (const float*)d_in[2];
  const float*         w  = (const float*)d_in[3];
  const float*         vy = (const float*)d_in[4];
  const int Ntot    = in_sizes[0] / 128;
  const int ngroups = (Ntot + 15) / 16;
  float* outX = (float*)d_out;
  float* outS = outX + (size_t)Ntot * 16;

  int nblocks = (ngroups + NW - 1) / NW;
  if (nblocks > 1042) nblocks = 1042;
  if (nblocks < 1) nblocks = 1;
  hipLaunchKernelGGL(ppca_estep, dim3(nblocks), dim3(NW * WSZ), SMEM_BYTES,
                     stream, Y, O, m, w, vy, outX, outS, Ntot, ngroups);
}
